// Attention_12077448036441
// MI455X (gfx1250) — matrix operand order repaired and checked by reading, not compiled, not run on hardware
//
#include <hip/hip_runtime.h>
#include <hip/hip_bf16.h>
#include <math.h>

// Problem constants (from reference)
#define ENC_DIM 2048
#define DEC_DIM 512
#define ATT_DIM 512
#define BATCH   512
#define NPIX    196
#define MROWS   (BATCH * NPIX)   // 100352 rows of the big GEMM, divisible by 64

typedef __attribute__((ext_vector_type(16))) __bf16        v16bf;
typedef __attribute__((ext_vector_type(8)))  float          v8f;
typedef __attribute__((ext_vector_type(16))) unsigned short v16us;
typedef __attribute__((ext_vector_type(8)))  unsigned short v8us;
typedef __attribute__((ext_vector_type(4)))  unsigned int   v4ui;
typedef __attribute__((ext_vector_type(8)))  unsigned int   v8ui;

// ---- environment probes (show up in stderr, harmless) ----------------------
#if defined(__gfx1250__)
#if __has_builtin(__builtin_amdgcn_tensor_load_to_lds)
#warning "probe: __builtin_amdgcn_tensor_load_to_lds PRESENT"
#else
#warning "probe: __builtin_amdgcn_tensor_load_to_lds ABSENT"
#endif
#if __has_builtin(__builtin_amdgcn_s_wait_tensorcnt)
#warning "probe: __builtin_amdgcn_s_wait_tensorcnt PRESENT"
#else
#warning "probe: __builtin_amdgcn_s_wait_tensorcnt ABSENT"
#endif
#if __clang_major__ >= 23
#warning "probe: clang >= 23"
#else
#warning "probe: clang < 23"
#endif
#endif

// TDM_MODE: 1 = issue B-tile DMA via inline-asm tensor_load_to_lds (portable
// across toolchains per CDNA5 bridge doc); 0 = manual vector-load staging.
#ifndef TDM_MODE
#define TDM_MODE 1
#endif
#if defined(__gfx1250__) && TDM_MODE
#define USE_TDM 1
#else
#define USE_TDM 0
#endif

__device__ __forceinline__ unsigned short f2bf(float f) {
    // round-to-nearest-even fp32 -> bf16
    unsigned int u = __float_as_uint(f);
    u += 0x7FFFu + ((u >> 16) & 1u);
    return (unsigned short)(u >> 16);
}

#if USE_TDM
__device__ __forceinline__ void tdm_wait0() {
#if __has_builtin(__builtin_amdgcn_s_wait_tensorcnt)
    __builtin_amdgcn_s_wait_tensorcnt(0);
#else
    asm volatile("s_wait_tensorcnt 0x0" ::: "memory");
#endif
}

// Issue a TDM 2D tile load: 512 rows (N) x 32 elements (K) of bf16 from
// W_enc^T (row stride ENC_DIM elements) into LDS at ldsAddr, N-major.
// D# per CDNA5 ISA 8.3/8.4: group0 {count=1, lds_addr, global_addr, type=2},
// group1 {data_size=2B, tensor_dim0=2048, tensor_dim1=512, tile_dim0=32,
// tile_dim1=512, tensor_dim0_stride=2048}; groups 2/3 zero (2D: tile_dim2/3=0
// => unused). Issued as inline asm so it works on both toolchains.
__device__ __forceinline__ void tdm_load_B(const unsigned short* src, unsigned ldsAddr) {
    unsigned long long ga = (unsigned long long)(const void*)src;
    v4ui g0 = (v4ui){ 1u,                                  // count=1, user desc
                      ldsAddr,                             // lds_addr[31:0]
                      (unsigned)ga,                        // global_addr[31:0]
                      (unsigned)(ga >> 32) | 0x80000000u };// ga[56:32] | type=2<<30
    v8ui g1 = (v8ui){ 0x00010000u,             // data_size = 1 (2 bytes)
                      2048u << 16,             // tensor_dim0 = 2048 (bits 79:48)
                      512u  << 16,             // tensor_dim1 = 512 (bits 111:80)
                      32u   << 16,             // tile_dim0 = 32 (bits 127:112)
                      512u,                    // tile_dim1 = 512 (bits 143:128)
                      2048u,                   // tensor_dim0_stride (bits 207:160)
                      0u, 0u };                // dim1_stride = 0 (2D)
    v4ui z4 = (v4ui){ 0u, 0u, 0u, 0u };
    asm volatile("tensor_load_to_lds %0, %1, %2, %3"
                 :: "s"(g0), "s"(g1), "s"(z4), "s"(z4)
                 : "memory");
}
#endif

// ---------------------------------------------------------------------------
// Kernel P: W_enc [E,A] fp32  ->  bf16 transposed [A,E] in workspace.
// N-major rows make WMMA B-fragment / TDM tile loads contiguous.
// ---------------------------------------------------------------------------
__global__ void prep_wenc_kernel(const float* __restrict__ W_enc,
                                 unsigned short* __restrict__ wT) {
    const int a = blockIdx.x;                       // 0..511
    #pragma unroll
    for (int i = 0; i < ENC_DIM / 256; ++i) {
        const int e = threadIdx.x + i * 256;
        wT[(size_t)a * ENC_DIM + e] = f2bf(W_enc[(size_t)e * ATT_DIM + a]);
    }
}

// ---------------------------------------------------------------------------
// Kernel 0: att2full[b,a] = decoder_hidden[b] @ W_dec + b_dec + b_enc
// (both biases folded here; tiny GEMM, 0.27 GFLOP)
// ---------------------------------------------------------------------------
__global__ void att2_kernel(const float* __restrict__ dh,
                            const float* __restrict__ W_dec,
                            const float* __restrict__ b_dec,
                            const float* __restrict__ b_enc,
                            float* __restrict__ att2) {
    __shared__ float h[DEC_DIM];
    const int b = blockIdx.x;
    h[threadIdx.x]       = dh[(size_t)b * DEC_DIM + threadIdx.x];
    h[threadIdx.x + 256] = dh[(size_t)b * DEC_DIM + threadIdx.x + 256];
    __syncthreads();
    for (int j = 0; j < 2; ++j) {
        const int a = threadIdx.x + j * 256;
        float acc = 0.f;
        for (int k = 0; k < DEC_DIM; ++k)
            acc = fmaf(h[k], W_dec[(size_t)k * ATT_DIM + a], acc);
        att2[(size_t)b * ATT_DIM + a] = acc + b_dec[a] + b_enc[a];
    }
}

// ---------------------------------------------------------------------------
// Kernel 1 (main): fused  scores = sum_a tanh(enc@W_enc + att2full) * W_full
// Block = 512 threads (16 wave32) handles 64 rows x all 512 att-dims.
// Wave w: row-tile (w&3)*16, N-quarter (w>>2)*128 -> 8 WMMA accum tiles.
// bf16 WMMA, f32 accumulate. B tiles DMA'd by the Tensor Data Mover into a
// ping-pong LDS buffer (next tile overlaps current compute); A converted
// fp32->bf16 through LDS by the VALU path.
// ---------------------------------------------------------------------------
__global__ __launch_bounds__(512) void scores_kernel(
        const float* __restrict__ enc,
        const unsigned short* __restrict__ wT,     // bf16 W_enc^T [A][E]
        const float* __restrict__ att2,
        const float* __restrict__ W_full,
        const float* __restrict__ b_full,
        float* __restrict__ scores) {
    __shared__ __align__(64) unsigned short ldsA[64 * 32];       // 4 KB
    __shared__ __align__(64) unsigned short ldsB[2][512 * 32];   // 2 x 32 KB
    __shared__ float          ldsS[64];

    const int tid     = threadIdx.x;
    const int wave    = tid >> 5;
    const int lane    = tid & 31;
    const int rt      = wave & 3;      // row tile within block (16 rows)
    const int nq      = wave >> 2;     // 128-wide N quarter
    const int rowBase = blockIdx.x * 64;

    if (tid < 64) ldsS[tid] = 0.f;

    v8f c[8] = {};                     // 8 accumulator tiles (64 VGPRs)

#if USE_TDM
    if (wave == 0)                     // prefetch first B tile via TDM
        tdm_load_B(wT, (unsigned)(size_t)(void*)&ldsB[0][0]);
#endif

    for (int kb = 0; kb < ENC_DIM; kb += 32) {
        const int sel = (kb >> 5) & 1;
        // --- stage A tile: 64 rows x 32 k, convert fp32 -> bf16 ---
        {
            const int row = tid >> 3;
            const int kk  = (tid & 7) * 4;
            const float4 v = *(const float4*)(enc + (size_t)(rowBase + row) * ENC_DIM + kb + kk);
            unsigned short* d = ldsA + row * 32 + kk;
            d[0] = f2bf(v.x); d[1] = f2bf(v.y); d[2] = f2bf(v.z); d[3] = f2bf(v.w);
        }
#if USE_TDM
        if (wave == 0)
            tdm_wait0();               // B tile for this kb landed
#else
        // --- fallback: stage B tile with vector loads ---
        #pragma unroll
        for (int i = 0; i < 4; ++i) {
            const int cIdx = tid + i * 512;    // 2048 8-elem chunks
            const int n    = cIdx >> 2;
            const int h    = cIdx & 3;
            *(uint4*)(&ldsB[sel][0] + n * 32 + h * 8) =
                *(const uint4*)(wT + (size_t)n * ENC_DIM + kb + h * 8);
        }
#endif
        __syncthreads();
#if USE_TDM
        if (wave == 0 && kb + 32 < ENC_DIM)    // overlap next DMA with compute
            tdm_load_B(wT + kb + 32, (unsigned)(size_t)(void*)&ldsB[sel ^ 1][0]);
#endif

        // A fragment (ISA 16-bit A 16x32 layout: lane<16 -> K{0..7,16..23},
        // lane>=16 -> K{8..15,24..31} of row M=lane&15)
        {
            const int m  = lane & 15;
            const int kh = lane >> 4;
            union { v16us u; v16bf b; v8us h2[2]; } ua;
            ua.h2[0] = *(const v8us*)(ldsA + (rt * 16 + m) * 32 + kh * 8);
            ua.h2[1] = *(const v8us*)(ldsA + (rt * 16 + m) * 32 + kh * 8 + 16);

            #pragma unroll
            for (int t = 0; t < 8; ++t) {
                // B fragment: lane<16 holds K{0..7,16..23}, lane>=16 holds K{8..15,24..31}
                // of column n0+(lane&15) -> two 16B LDS reads.
                const int n0 = nq * 128 + t * 16;
                union { v16us u; v16bf b; v8us h2[2]; } ub;
                ub.h2[0] = *(const v8us*)(&ldsB[sel][0] + (n0 + m) * 32 + kh * 8);
                ub.h2[1] = *(const v8us*)(&ldsB[sel][0] + (n0 + m) * 32 + kh * 8 + 16);
                c[t] = __builtin_amdgcn_wmma_f32_16x16x32_bf16(
                    false, ua.b, false, ub.b, (short)0, c[t], false, false);
            }
        }
        __syncthreads();
    }

    // --- fused epilogue: tanh(att1 + att2) * W_full, reduce over att dim ---
    // C layout: VGPR i holds M=i (lanes 0-15) / M=i+8 (lanes 16-31), N=lane&15.
    const int m  = lane & 15;
    const int mo = (lane >> 4) * 8;
    int bi[8];
    #pragma unroll
    for (int i = 0; i < 8; ++i)
        bi[i] = (rowBase + rt * 16 + mo + i) / NPIX;   // batch of each row

    float partial[8];
    #pragma unroll
    for (int i = 0; i < 8; ++i) partial[i] = 0.f;

    #pragma unroll
    for (int t = 0; t < 8; ++t) {
        const int   n  = nq * 128 + t * 16 + m;
        const float wf = W_full[n];
        #pragma unroll
        for (int i = 0; i < 8; ++i) {
            const float v = c[t][i] + att2[(size_t)bi[i] * ATT_DIM + n];
            partial[i] += tanhf(v) * wf;
        }
    }
    #pragma unroll
    for (int i = 0; i < 8; ++i)
        atomicAdd(&ldsS[rt * 16 + mo + i], partial[i]);   // ds_add_f32
    __syncthreads();

    if (tid < 64)
        scores[(size_t)rowBase + tid] = ldsS[tid] + b_full[0];
}

// ---------------------------------------------------------------------------
// Kernel 2: softmax over P=196 per batch
// ---------------------------------------------------------------------------
__global__ void softmax_kernel(const float* __restrict__ scores,
                               float* __restrict__ alpha) {
    __shared__ float red[256];
    const int b = blockIdx.x, tid = threadIdx.x;
    const float v = (tid < NPIX) ? scores[(size_t)b * NPIX + tid] : -3.0e38f;
    red[tid] = v;
    __syncthreads();
    for (int s = 128; s > 0; s >>= 1) {
        if (tid < s) red[tid] = fmaxf(red[tid], red[tid + s]);
        __syncthreads();
    }
    const float mx = red[0];
    __syncthreads();
    const float e = (tid < NPIX) ? __expf(v - mx) : 0.f;
    red[tid] = e;
    __syncthreads();
    for (int s = 128; s > 0; s >>= 1) {
        if (tid < s) red[tid] += red[tid + s];
        __syncthreads();
    }
    const float inv = 1.f / red[0];
    if (tid < NPIX) alpha[(size_t)b * NPIX + tid] = e * inv;
}

// ---------------------------------------------------------------------------
// Kernel 3: context[b,e] = sum_p enc[b,p,e] * alpha[b,p]  (HBM-bound, float4)
// ---------------------------------------------------------------------------
__global__ void context_kernel(const float* __restrict__ enc,
                               const float* __restrict__ alpha,
                               float* __restrict__ ctx) {
    __shared__ float al[NPIX];
    const int b = blockIdx.x, tid = threadIdx.x;
    if (tid < NPIX) al[tid] = alpha[(size_t)b * NPIX + tid];
    __syncthreads();
    const int e4 = blockIdx.y * 256 + tid;                 // float4 index (0..511)
    const float4* ep = (const float4*)(enc + (size_t)b * NPIX * ENC_DIM);
    float4 acc = {0.f, 0.f, 0.f, 0.f};
    for (int p = 0; p < NPIX; ++p) {
        const float4 x = ep[(size_t)p * (ENC_DIM / 4) + e4];
        const float a = al[p];
        acc.x = fmaf(x.x, a, acc.x);
        acc.y = fmaf(x.y, a, acc.y);
        acc.z = fmaf(x.z, a, acc.z);
        acc.w = fmaf(x.w, a, acc.w);
    }
    ((float4*)(ctx + (size_t)b * ENC_DIM))[e4] = acc;
}

// ---------------------------------------------------------------------------
extern "C" void kernel_launch(void* const* d_in, const int* in_sizes, int n_in,
                              void* d_out, int out_size, void* d_ws, size_t ws_size,
                              hipStream_t stream) {
    const float* enc   = (const float*)d_in[0];
    const float* dh    = (const float*)d_in[1];
    const float* Wenc  = (const float*)d_in[2];
    const float* benc  = (const float*)d_in[3];
    const float* Wdec  = (const float*)d_in[4];
    const float* bdec  = (const float*)d_in[5];
    const float* Wfull = (const float*)d_in[6];
    const float* bfull = (const float*)d_in[7];

    // workspace: bf16 W_enc^T (2 MB) | att2full (1 MB) | scores (0.4 MB)
    char* ws = (char*)d_ws;
    unsigned short* wT = (unsigned short*)ws;
    float* att2   = (float*)(ws + (size_t)ATT_DIM * ENC_DIM * sizeof(unsigned short));
    float* scores = (float*)(ws + (size_t)ATT_DIM * ENC_DIM * sizeof(unsigned short)
                                + (size_t)BATCH * ATT_DIM * sizeof(float));

    float* ctx   = (float*)d_out;                              // [B, E]
    float* alpha = (float*)d_out + (size_t)BATCH * ENC_DIM;    // [B, P]

    prep_wenc_kernel<<<ATT_DIM, 256, 0, stream>>>(Wenc, wT);
    att2_kernel<<<BATCH, 256, 0, stream>>>(dh, Wdec, bdec, benc, att2);
    scores_kernel<<<MROWS / 64, 512, 0, stream>>>(enc, wT, att2, Wfull, bfull, scores);
    softmax_kernel<<<BATCH, 256, 0, stream>>>(scores, alpha);
    context_kernel<<<dim3(BATCH, ENC_DIM / 1024), 256, 0, stream>>>(enc, alpha, ctx);
}
